// RGMU_68478958567500
// MI455X (gfx1250) — matrix operand order repaired and checked by reading, not compiled, not run on hardware
//
#include <hip/hip_runtime.h>
#include <hip/hip_bf16.h>

#define HASB(x) __has_builtin(x)

typedef __attribute__((ext_vector_type(16))) __bf16 bf16x16;
typedef __attribute__((ext_vector_type(8)))  float  f32x8;
typedef __attribute__((ext_vector_type(4)))  unsigned int u32x4;
typedef __attribute__((ext_vector_type(8)))  int    i32x8;
typedef __attribute__((ext_vector_type(4)))  int    i32x4;

namespace {
constexpr int kT = 512, kC = 4, kF = 128, kO = 128;
constexpr int kK = kC * kF;               // 512 (flat inner dim, == C*O too)
constexpr int ROWS = 16;                  // batch rows per workgroup
// LDS layouts
constexpr int XSTRIDE = 2080;             // fp32 x row: 512*4 bytes + 2 pads of 16B (TDM pad: 4dw per 256dw)
constexpr int SSTRIDE = 1040;             // bf16 state row: 520 elems -> 4-bank skew, conflict free
constexpr int XBUFB = ROWS * XSTRIDE;     // 33280
constexpr int SBUFB = ROWS * SSTRIDE;     // 16640
constexpr int LDS_X0 = 0;
constexpr int LDS_X1 = XBUFB;
constexpr int LDS_H0 = 2 * XBUFB;
constexpr int LDS_Z0 = LDS_H0 + 2 * SBUFB;
constexpr int LDS_TOTAL = 2 * XBUFB + 4 * SBUFB;  // 133120 bytes (< 320KB WGP LDS)
// bf16 weight-fragment arena offsets inside d_ws (elements)
constexpr size_t WZ_OFF  = 0;
constexpr size_t WZL_OFF = (size_t)512 * 512;
constexpr size_t WH_OFF  = (size_t)2 * 512 * 512;
constexpr size_t WHL_OFF = WH_OFF + (size_t)4 * 128 * 128;
}

// ---------------------------------------------------------------------------
// Prep: pack a row-major fp32 KxN weight into bf16 B-fragments for
// v_wmma_f32_16x16x32_bf16.  B tile is 32x16; per ISA 7.12.2 lanes 0-15 hold
// K=0..7,16..23 (j=0..7, 8..15), lanes 16-31 hold K=8..15,24..31.
// ---------------------------------------------------------------------------
__global__ void pack_bfrag(const float* __restrict__ src, __bf16* __restrict__ dst,
                           int K, int Ncols) {
  int id = blockIdx.x * blockDim.x + threadIdx.x;
  int total = K * Ncols;
  if (id >= total) return;
  int j    = id & 15;
  int lane = (id >> 4) & 31;
  int NT   = Ncols >> 4;
  int nt   = (id >> 9) % NT;
  int kt   = (id >> 9) / NT;
  int k = kt * 32 + ((lane >> 4) << 3) + (j & 7) + ((j >> 3) << 4);   // lanes 16-31 -> K += 8; j>=8 -> K += 16
  int n = nt * 16 + (lane & 15);
  dst[id] = (__bf16)src[(size_t)k * Ncols + n];
}

// ---------------------------------------------------------------------------
// Fragment builders (A layout per ISA 7.12.2 16-bit A 16x32).
// ---------------------------------------------------------------------------
__device__ __forceinline__ bf16x16 make_xfrag(const char* xrow_base, int kt, int lane) {
  int row = lane & 15;
  int k0  = kt * 32 + ((lane >> 4) << 3);
  int pad = (kt >= 8) ? 16 : 0;                       // TDM pad after dword 256
  const char* a0 = xrow_base + row * XSTRIDE + k0 * 4 + pad;
  float4 f0 = *(const float4*)(a0);
  float4 f1 = *(const float4*)(a0 + 16);
  float4 f2 = *(const float4*)(a0 + 64);              // k0+16
  float4 f3 = *(const float4*)(a0 + 80);
  bf16x16 v;
  v[0]=(__bf16)f0.x; v[1]=(__bf16)f0.y; v[2]=(__bf16)f0.z; v[3]=(__bf16)f0.w;
  v[4]=(__bf16)f1.x; v[5]=(__bf16)f1.y; v[6]=(__bf16)f1.z; v[7]=(__bf16)f1.w;
  v[8]=(__bf16)f2.x; v[9]=(__bf16)f2.y; v[10]=(__bf16)f2.z; v[11]=(__bf16)f2.w;
  v[12]=(__bf16)f3.x; v[13]=(__bf16)f3.y; v[14]=(__bf16)f3.z; v[15]=(__bf16)f3.w;
  return v;
}

__device__ __forceinline__ bf16x16 make_sfrag(const char* sbuf, int kt, int lane) {
  int row = lane & 15;
  int k0  = kt * 32 + ((lane >> 4) << 3);
  const char* a = sbuf + row * SSTRIDE + k0 * 2;
  union { struct { uint4 lo, hi; } raw; bf16x16 v; } u;
  u.raw.lo = *(const uint4*)(a);        // K k0..k0+7
  u.raw.hi = *(const uint4*)(a + 32);   // K k0+16..k0+23
  return u.v;
}

__device__ __forceinline__ f32x8 wmma_bf16(bf16x16 a, bf16x16 b, f32x8 c) {
  return __builtin_amdgcn_wmma_f32_16x16x32_bf16(false, a, false, b, (short)0, c, false, false);
}

__device__ __forceinline__ float fast_tanh(float x) {
#if HASB(__builtin_amdgcn_tanhf)
  return __builtin_amdgcn_tanhf(x);
#else
  return tanhf(x);
#endif
}

// Opaque uniform zero, regenerated every loop iteration.  Added into the
// (typed, global) weight pointers it makes the B-fragment addresses opaque to
// LICM -- the loads stay inside the time loop (no 160KB/wave scratch spill) --
// while keeping addrspace(1) so they lower to global_load_b128 with the
// fragment offsets folded into the 24-bit instruction immediate.
__device__ __forceinline__ unsigned opaque_zero() {
  unsigned z = 0;
  asm volatile("" : "+s"(z));
  return z;
}

// ---------------------------------------------------------------------------
// One GEMM phase: acc[i] += A(16x32 tiles from LDS) x B(frags from L2 arena).
// Depth-1 software pipeline with static ping-pong registers (no copies):
// B-frags for kt+1 are issued before the WMMAs of kt so L2 latency is covered
// by 4 WMMAs + the LDS A-build.  wbase already includes nt/lane offsets, so
// every load is base + compile-time-constant immediate.
// ---------------------------------------------------------------------------
template <int KTILES, int NT, bool ISX>
__device__ __forceinline__ void gemm_phase(const __bf16* wbase, const char* abuf,
                                           int ktbase, int lane, f32x8 acc[4]) {
  bf16x16 buf[2][4];
#pragma unroll
  for (int i = 0; i < 4; ++i) buf[0][i] = *(const bf16x16*)(wbase + i * 512);
#pragma unroll
  for (int kt = 0; kt < KTILES; ++kt) {
    const int cur = kt & 1, nxt = cur ^ 1;
    if (kt + 1 < KTILES) {
#pragma unroll
      for (int i = 0; i < 4; ++i)
        buf[nxt][i] = *(const bf16x16*)(wbase + (size_t)(kt + 1) * (NT * 512) + i * 512);
    }
    bf16x16 a = ISX ? make_xfrag(abuf, ktbase + kt, lane)
                    : make_sfrag(abuf, ktbase + kt, lane);
#pragma unroll
    for (int i = 0; i < 4; ++i) acc[i] = wmma_bf16(a, buf[cur][i], acc[i]);
  }
}

// ---------------------------------------------------------------------------
// TDM: load x[rowbase..rowbase+15, t, :, :] (16 rows x 2KB, row stride 1MB)
// into LDS with padding (interval=256 dwords, amount=4 dwords) -> XSTRIDE.
// ---------------------------------------------------------------------------
#if HASB(__builtin_amdgcn_tensor_load_to_lds)
#define USE_TDM 1
__device__ __forceinline__ void tdm_load_x(const float* x, int rowbase, int t, unsigned lds_off) {
  unsigned long long ga =
      (unsigned long long)(const void*)(x + (size_t)rowbase * (kT * kK) + (size_t)t * kK);
  u32x4 g0;
  g0[0] = 1u;                                    // count=1, no gather
  g0[1] = lds_off;                               // lds_addr
  g0[2] = (unsigned)(ga & 0xFFFFFFFFu);          // global_addr[31:0]
  g0[3] = (unsigned)((ga >> 32) & 0x01FFFFFFu) | (2u << 30);  // addr[56:32] | type=2
  i32x8 g1;
  g1[0] = (2 << 16) | (1 << 20) | (7 << 22) | (3 << 25);  // 4B elems, pad_en, 256dw interval, 4dw pad
  g1[1] = (int)(512u << 16);                     // tensor_dim0 = 512
  g1[2] = (int)(16u << 16);                      // tensor_dim1 = 16
  g1[3] = (int)(512u << 16);                     // tile_dim0 = 512
  g1[4] = 16;                                    // tile_dim1 = 16, tile_dim2 = 0
  g1[5] = kT * kK;                               // tensor_dim0_stride = 262144 elems
  g1[6] = 0;
  g1[7] = 0;
  i32x4 gz = {0, 0, 0, 0};                       // groups 2/3 unused (2D tensor)
  i32x8 gx = {0, 0, 0, 0, 0, 0, 0, 0};           // extra group (unused)
  __builtin_amdgcn_tensor_load_to_lds(g0, g1, gz, gz, gx, 0);
}
#else
#define USE_TDM 0
#endif

// ---------------------------------------------------------------------------
// Main persistent recurrent kernel: 8 blocks x 256 threads (8 wave32).
// Wave w owns output column tiles nt = 4w..4w+3 (of 32 tiles over C*O=512).
// ---------------------------------------------------------------------------
__global__ __launch_bounds__(256, 1)
void rgmu_kernel(const float* __restrict__ x, const __bf16* __restrict__ wf,
                 const float* __restrict__ b_h, const float* __restrict__ b_z,
                 float* __restrict__ out) {
  extern __shared__ char smem[];
  const int tid  = threadIdx.x;
  const int lane = tid & 31;
  const int wv   = __builtin_amdgcn_readfirstlane(tid >> 5);  // wave-uniform in SGPR
  const int rowbase = blockIdx.x * ROWS;

  // zero the state arena (H0,H1,Z0,Z1); t=0 reads buffer parity 1 which must be 0
  {
    unsigned* st = (unsigned*)(smem + LDS_H0);
    for (int i = tid; i < (4 * SBUFB) >> 2; i += 256) st[i] = 0u;
  }

  // stage x(t=0) into X0
#if USE_TDM
  if (wv == 0) {
    tdm_load_x(x, rowbase, 0, (unsigned)(size_t)(smem + LDS_X0));
    __builtin_amdgcn_s_wait_tensorcnt(0);
  }
#else
  {
    int row = tid >> 4, seg = tid & 15;
    const float* p = x + (size_t)(rowbase + row) * (kT * kK) + seg * 32;
#pragma unroll
    for (int i = 0; i < 8; ++i) {
      int k = seg * 32 + i * 4;
      *(float4*)(smem + LDS_X0 + row * XSTRIDE + k * 4 + (k >= 256 ? 16 : 0)) =
          ((const float4*)p)[i];
    }
  }
#endif
  __syncthreads();

  const int nt0  = wv * 4;            // first of 4 column tiles
  const int c    = wv >> 1;           // modality of this wave's tiles
  const int ntl0 = (wv & 1) * 4;      // tile index within the modality's 8 tiles

  // per-wave/lane element offsets inside each fragment arena (constant over t)
  const int zoff = nt0 * 512 + lane * 16;                 // W_z / W_zl arenas (NT=32)
  const int hoff = c * 16384 + ntl0 * 512 + lane * 16;    // W_h / W_hl arenas (NT=8, per-c)

  float bh[4], bz[4];
#pragma unroll
  for (int i = 0; i < 4; ++i) {
    int ogl = (nt0 + i) * 16 + (lane & 15);   // flat (c*O+o) column, == bias index
    bh[i] = b_h[ogl];
    bz[i] = b_z[ogl];
  }

  const f32x8 vzero = {0.f, 0.f, 0.f, 0.f, 0.f, 0.f, 0.f, 0.f};
  const int row0 = (lane >> 4) << 3;

  for (int t = 0; t < kT; ++t) {
    const int cb = t & 1, pb = cb ^ 1;
    const char* xcur  = smem + (cb ? LDS_X1 : LDS_X0);
    const char* hprev = smem + LDS_H0 + pb * SBUFB;
    const char* zprev = smem + LDS_Z0 + pb * SBUFB;
    char* hcur = smem + LDS_H0 + cb * SBUFB;
    char* zcur = smem + LDS_Z0 + cb * SBUFB;

    // opaque-per-iteration zero keeps the B-fragment loads inside the t loop
    const unsigned zr = opaque_zero();
    const __bf16* wz_w  = wf + (WZ_OFF  + zr) + zoff;
    const __bf16* wzl_w = wf + (WZL_OFF + zr) + zoff;
    const __bf16* wh_w  = wf + (WH_OFF  + zr) + hoff;
    const __bf16* whl_w = wf + (WHL_OFF + zr) + hoff;

    // ---- prefetch x(t+1) into the other buffer (overlaps the WMMA work) ----
#if USE_TDM
    if (wv == 0 && t + 1 < kT)
      tdm_load_x(x, rowbase, t + 1, (unsigned)(size_t)(smem + (pb ? LDS_X1 : LDS_X0)));
#else
    float4 xr[8];
    if (t + 1 < kT) {
      int row = tid >> 4, seg = tid & 15;
      const float* p = x + (size_t)(rowbase + row) * (kT * kK) + (size_t)(t + 1) * kK + seg * 32;
#pragma unroll
      for (int i = 0; i < 8; ++i) xr[i] = ((const float4*)p)[i];
    }
#endif

    f32x8 accZ[4] = {vzero, vzero, vzero, vzero};
    f32x8 accH[4] = {vzero, vzero, vzero, vzero};

    // gate path: x_t(16x512) @ W_z  +  z_prev(16x512) @ W_zl
    gemm_phase<16, 32, true >(wz_w,  xcur,  0, lane, accZ);
    gemm_phase<16, 32, false>(wzl_w, zprev, 0, lane, accZ);
    // hidden path (block diagonal, modality c): x_c @ W_h[c] + h_c @ W_hl[c]
    gemm_phase<4,  8,  true >(wh_w,  xcur,  c * 4, lane, accH);
    gemm_phase<4,  8,  false>(whl_w, hprev, c * 4, lane, accH);

    // ---- epilogue: activations, write bf16 state (A-operand row-major layout) ----
#pragma unroll
    for (int i = 0; i < 4; ++i) {
      int coloff = ((nt0 + i) * 16 + (lane & 15)) * 2;
#pragma unroll
      for (int r = 0; r < 8; ++r) {
        float hv = fast_tanh(accH[i][r] + bh[i]);
        float zl = accZ[i][r] + bz[i];
        float zv = 1.0f / (1.0f + __expf(-zl));
        *(__bf16*)(hcur + (row0 + r) * SSTRIDE + coloff) = (__bf16)hv;
        *(__bf16*)(zcur + (row0 + r) * SSTRIDE + coloff) = (__bf16)zv;
      }
    }

    // ---- commit x(t+1); make everything visible; ping-pong ----
#if USE_TDM
    if (wv == 0) __builtin_amdgcn_s_wait_tensorcnt(0);
#else
    if (t + 1 < kT) {
      char* xnext = smem + (pb ? LDS_X1 : LDS_X0);
      int row = tid >> 4, seg = tid & 15;
#pragma unroll
      for (int i = 0; i < 8; ++i) {
        int k = seg * 32 + i * 4;
        *(float4*)(xnext + row * XSTRIDE + k * 4 + (k >= 256 ? 16 : 0)) = xr[i];
      }
    }
#endif
    __syncthreads();
  }

  // ---- fuse: out[n,o] = sum_c h[n,c,o] * z[n,c,o] ----
  const char* hf = smem + LDS_H0 + ((kT - 1) & 1) * SBUFB;
  const char* zf = smem + LDS_Z0 + ((kT - 1) & 1) * SBUFB;
  for (int idx = tid; idx < ROWS * kO; idx += 256) {
    int row = idx >> 7, o = idx & 127;
    float s = 0.f;
#pragma unroll
    for (int cc = 0; cc < kC; ++cc) {
      float hv = (float)*(const __bf16*)(hf + row * SSTRIDE + (cc * 128 + o) * 2);
      float zv = (float)*(const __bf16*)(zf + row * SSTRIDE + (cc * 128 + o) * 2);
      s += hv * zv;
    }
    out[(size_t)(rowbase + row) * kO + o] = s;
  }
}

// ---------------------------------------------------------------------------
extern "C" void kernel_launch(void* const* d_in, const int* in_sizes, int n_in,
                              void* d_out, int out_size, void* d_ws, size_t ws_size,
                              hipStream_t stream) {
  (void)in_sizes; (void)n_in; (void)out_size; (void)ws_size;
  const float* x    = (const float*)d_in[0];
  const float* W_h  = (const float*)d_in[1];
  const float* W_hl = (const float*)d_in[2];
  const float* W_z  = (const float*)d_in[3];
  const float* W_zl = (const float*)d_in[4];
  const float* b_h  = (const float*)d_in[5];
  const float* b_z  = (const float*)d_in[6];
  __bf16* wsb = (__bf16*)d_ws;

  // pack weights -> bf16 B-fragments (L2 resident for the whole recurrence)
  pack_bfrag<<<(512 * 512 + 255) / 256, 256, 0, stream>>>(W_z,  wsb + WZ_OFF,  512, 512);
  pack_bfrag<<<(512 * 512 + 255) / 256, 256, 0, stream>>>(W_zl, wsb + WZL_OFF, 512, 512);
  for (int c = 0; c < 4; ++c) {
    pack_bfrag<<<(128 * 128 + 255) / 256, 256, 0, stream>>>(
        W_h + (size_t)c * 128 * 128, wsb + WH_OFF + (size_t)c * 128 * 128, 128, 128);
    pack_bfrag<<<(128 * 128 + 255) / 256, 256, 0, stream>>>(
        W_hl + (size_t)c * 128 * 128, wsb + WHL_OFF + (size_t)c * 128 * 128, 128, 128);
  }

  static_assert(LDS_TOTAL <= 160 * 1024, "LDS budget");
  (void)hipFuncSetAttribute((const void*)rgmu_kernel,
                            hipFuncAttributeMaxDynamicSharedMemorySize, LDS_TOTAL);
  rgmu_kernel<<<128 / ROWS, 256, LDS_TOTAL, stream>>>(x, wsb, b_h, b_z, (float*)d_out);
}
